// EntangledInterferenceLayer_18150531793283
// MI455X (gfx1250) — matrix operand order repaired and checked by reading, not compiled, not run on hardware
//
#include <hip/hip_runtime.h>
#include <math.h>

// ---------------------------------------------------------------------------
// EntangledInterferenceLayer for MI455X (gfx1250, wave32, WMMA).
// All matmuls (8 projections + QK^T + PV) go through v_wmma_f32_16x16x32_bf16.
// ---------------------------------------------------------------------------

typedef __attribute__((ext_vector_type(16))) __bf16         v16bf;
typedef __attribute__((ext_vector_type(8)))  float          v8f;
typedef __attribute__((ext_vector_type(8)))  unsigned short ushort8;
typedef __attribute__((ext_vector_type(16))) unsigned short ushort16;

#define DEV __device__ __forceinline__

static constexpr int kDim = 1024;
static constexpr int kS   = 1024;
static constexpr int kB   = 2;
static constexpr int kH   = 16;
static constexpr int kHD  = 64;
static constexpr int kTok = kB * kS;   // 2048 rows for the GEMMs

DEV unsigned short f32_to_bf16(float f) {
  unsigned int u = __float_as_uint(f);
  u += 0x7fffu + ((u >> 16) & 1u);     // round-to-nearest-even
  return (unsigned short)(u >> 16);
}

DEV v8f wmma_bf16(v16bf a, v16bf b, v8f c) {
  // 8 args: (neg_a, A, neg_b, B, c_mod, C, reuse_a, reuse_b)
  return __builtin_amdgcn_wmma_f32_16x16x32_bf16(false, a, false, b,
                                                 (short)0, c, false, false);
}

// A fragment (16x32 bf16, row-major source, ld = ldk elements).
// Layout used here: lane L -> row M = L%16; elems 0..15 = K(k0 + (L<16?0:16) + i)
// (contiguous, matching load_b_frag).
DEV v16bf load_a_frag(const unsigned short* A, int row0, int k0, int ldk) {
  int lane  = threadIdx.x & 31;
  int half8 = (lane < 16) ? 0 : 16;
  const unsigned short* p = A + (size_t)(row0 + (lane & 15)) * ldk + k0 + half8;
  ushort8 lo = *(const ushort8*)(p);
  ushort8 hi = *(const ushort8*)(p + 8);
  ushort16 v = __builtin_shufflevector(lo, hi,
      0,1,2,3,4,5,6,7,8,9,10,11,12,13,14,15);
  return __builtin_bit_cast(v16bf, v);
}

// B fragment (32x16 bf16) from a K-major (transposed) matrix BT[N][K].
// Layout used here: lane L -> col N = L%16; contiguous K = k0 + (L<16?0:16) + 0..15.
DEV v16bf load_b_frag(const unsigned short* BT, int col0, int k0, int ldk) {
  int lane = threadIdx.x & 31;
  const unsigned short* p =
      BT + (size_t)(col0 + (lane & 15)) * ldk + k0 + ((lane < 16) ? 0 : 16);
  ushort16 v = *(const ushort16*)(p);
  return __builtin_bit_cast(v16bf, v);
}

DEV v16bf neg_frag(v16bf a) {   // bf16 NEG field is C-only per ISA; flip signs manually
  ushort16 u = __builtin_bit_cast(ushort16, a);
#pragma unroll
  for (int i = 0; i < 16; ++i) u[i] ^= (unsigned short)0x8000;
  return __builtin_bit_cast(v16bf, u);
}

// ---------------------------------------------------------------------------
// fp32 -> bf16 convert
__global__ __launch_bounds__(256) void cvt_kernel(const float* __restrict__ x,
                                                  unsigned short* __restrict__ y,
                                                  int n) {
  int i = blockIdx.x * blockDim.x + threadIdx.x;
  if (i < n) y[i] = f32_to_bf16(x[i]);
}

// W[K][N] fp32 -> WT[N][K] bf16 (so B fragments are contiguous lane loads)
__global__ __launch_bounds__(256) void wtrans_kernel(const float* __restrict__ W,
                                                     unsigned short* __restrict__ WT) {
  __shared__ float tile[32][33];
  int n0 = blockIdx.x * 32, k0 = blockIdx.y * 32;
  int tx = threadIdx.x & 31, ty = threadIdx.x >> 5;   // 8 row-groups
  for (int i = ty; i < 32; i += 8)
    tile[i][tx] = W[(size_t)(k0 + i) * kDim + n0 + tx];
  __syncthreads();
  for (int i = ty; i < 32; i += 8)
    WT[(size_t)(n0 + i) * kDim + k0 + tx] = f32_to_bf16(tile[tx][i]);
}

// ---------------------------------------------------------------------------
// C[M][N] fp32 = A_bf16[M][K] * BT_bf16[N][K]^T + bias[N]
// 8 waves / block; each wave owns a 32(M) x 64(N) tile -> block tile 128x128.
__global__ __launch_bounds__(256) void gemm_bf16_kernel(
    const unsigned short* __restrict__ A, const unsigned short* __restrict__ BT,
    const float* __restrict__ bias, float* __restrict__ C,
    int M, int N, int K) {
  int wave = threadIdx.x >> 5, lane = threadIdx.x & 31;
  int mw = wave & 3, nw = wave >> 2;
  int mBase = blockIdx.x * 128 + mw * 32;
  int nBase = blockIdx.y * 128 + nw * 64;
  v8f acc[2][4] = {};
  for (int k0 = 0; k0 < K; k0 += 32) {
    __builtin_prefetch(A + (size_t)(mBase + (lane & 15)) * K + k0 + 64, 0, 1);
    v16bf a0 = load_a_frag(A, mBase,      k0, K);
    v16bf a1 = load_a_frag(A, mBase + 16, k0, K);
#pragma unroll
    for (int j = 0; j < 4; ++j) {
      v16bf bb = load_b_frag(BT, nBase + j * 16, k0, K);
      acc[0][j] = wmma_bf16(a0, bb, acc[0][j]);
      acc[1][j] = wmma_bf16(a1, bb, acc[1][j]);
    }
  }
  // C/D layout: lane L -> col N=L%16; VGPR r -> row r + (L<16?0:8)
  int nCol = lane & 15, rowOff = (lane < 16) ? 0 : 8;
#pragma unroll
  for (int i = 0; i < 2; ++i)
#pragma unroll
    for (int j = 0; j < 4; ++j) {
      int n = nBase + j * 16 + nCol;
      float bv = bias[n];
#pragma unroll
      for (int r = 0; r < 8; ++r) {
        int mrow = mBase + i * 16 + r + rowOff;
        C[(size_t)mrow * N + n] = acc[i][j][r] + bv;
      }
    }
}

// ---------------------------------------------------------------------------
// v fp32 [B,S,H,D] -> Vt bf16 [B,H,D,S]  (so PV B-fragments are contiguous)
__global__ __launch_bounds__(256) void v_transpose_kernel(
    const float* __restrict__ v, unsigned short* __restrict__ Vt) {
  int idx = blockIdx.x * blockDim.x + threadIdx.x;
  if (idx >= kB * kS * kDim) return;
  int d = idx & (kHD - 1);
  int h = (idx >> 6) & (kH - 1);
  int s = (idx >> 10) & (kS - 1);
  int b = idx >> 20;
  Vt[(((size_t)(b * kH + h)) * kHD + d) * kS + s] = f32_to_bf16(v[idx]);
}

// ---------------------------------------------------------------------------
// RoPE (first 32 dims) -> ent head-mix -> per-head complex phase shift.
// One thread per (b, s, dim-pair p); handles all 16 heads for that pair.
__global__ __launch_bounds__(256) void rope_mix_phase_kernel(
    const float* __restrict__ qr, const float* __restrict__ qi,
    const float* __restrict__ kr, const float* __restrict__ ki,
    const float* __restrict__ phase, const float* __restrict__ ent,
    const float* __restrict__ rotf,
    unsigned short* __restrict__ Qr, unsigned short* __restrict__ Qi,
    unsigned short* __restrict__ Kr, unsigned short* __restrict__ Ki) {
  int idx = blockIdx.x * blockDim.x + threadIdx.x;
  if (idx >= kB * kS * (kHD / 2)) return;
  int p = idx & 31;
  int s = (idx >> 5) & (kS - 1);
  int b = idx >> 15;
  int d0 = 2 * p, d1 = d0 + 1;
  float c = 1.f, sn = 0.f;
  bool rot = (d0 < 32);
  if (rot) { float ang = (float)s * rotf[p]; c = __cosf(ang); sn = __sinf(ang); }

  float Aqr0[kH], Aqr1[kH], Aqi0[kH], Aqi1[kH];
  float Akr0[kH], Akr1[kH], Aki0[kH], Aki1[kH];
  size_t base = ((size_t)(b * kS + s)) * kDim;
#pragma unroll
  for (int h = 0; h < kH; ++h) {
    size_t o0 = base + h * kHD + d0;
    float x0, x1;
    x0 = qr[o0]; x1 = qr[o0 + 1];
    Aqr0[h] = rot ? x0 * c - x1 * sn : x0;  Aqr1[h] = rot ? x1 * c + x0 * sn : x1;
    x0 = qi[o0]; x1 = qi[o0 + 1];
    Aqi0[h] = rot ? x0 * c - x1 * sn : x0;  Aqi1[h] = rot ? x1 * c + x0 * sn : x1;
    x0 = kr[o0]; x1 = kr[o0 + 1];
    Akr0[h] = rot ? x0 * c - x1 * sn : x0;  Akr1[h] = rot ? x1 * c + x0 * sn : x1;
    x0 = ki[o0]; x1 = ki[o0 + 1];
    Aki0[h] = rot ? x0 * c - x1 * sn : x0;  Aki1[h] = rot ? x1 * c + x0 * sn : x1;
  }
#pragma unroll 1
  for (int x = 0; x < kH; ++x) {
    float mqr0 = 0, mqr1 = 0, mqi0 = 0, mqi1 = 0;
    float mkr0 = 0, mkr1 = 0, mki0 = 0, mki1 = 0;
#pragma unroll
    for (int h2 = 0; h2 < kH; ++h2) {
      float e = ent[h2 * kH + x];          // out[x] = sum_h in[h]*ent[h][x]
      mqr0 += Aqr0[h2] * e; mqr1 += Aqr1[h2] * e;
      mqi0 += Aqi0[h2] * e; mqi1 += Aqi1[h2] * e;
      mkr0 += Akr0[h2] * e; mkr1 += Akr1[h2] * e;
      mki0 += Aki0[h2] * e; mki1 += Aki1[h2] * e;
    }
    float pc0 = __cosf(phase[x * kHD + d0]), ps0 = __sinf(phase[x * kHD + d0]);
    float pc1 = __cosf(phase[x * kHD + d1]), ps1 = __sinf(phase[x * kHD + d1]);
    float oqr0 = mqr0 * pc0 - mqi0 * ps0, oqi0 = mqr0 * ps0 + mqi0 * pc0;
    float oqr1 = mqr1 * pc1 - mqi1 * ps1, oqi1 = mqr1 * ps1 + mqi1 * pc1;
    float okr0 = mkr0 * pc0 - mki0 * ps0, oki0 = mkr0 * ps0 + mki0 * pc0;
    float okr1 = mkr1 * pc1 - mki1 * ps1, oki1 = mkr1 * ps1 + mki1 * pc1;
    size_t ob = (((size_t)(b * kH + x)) * kS + s) * kHD;
    Qr[ob + d0] = f32_to_bf16(oqr0); Qr[ob + d1] = f32_to_bf16(oqr1);
    Qi[ob + d0] = f32_to_bf16(oqi0); Qi[ob + d1] = f32_to_bf16(oqi1);
    Kr[ob + d0] = f32_to_bf16(okr0); Kr[ob + d1] = f32_to_bf16(okr1);
    Ki[ob + d0] = f32_to_bf16(oki0); Ki[ob + d1] = f32_to_bf16(oki1);
  }
}

// ---------------------------------------------------------------------------
// Flash-style causal attention on complex score magnitude.
// grid = (S/128, B*H); 8 waves/block, each wave owns 16 query rows.
__global__ __launch_bounds__(256) void attention_kernel(
    const unsigned short* __restrict__ Qr, const unsigned short* __restrict__ Qi,
    const unsigned short* __restrict__ Kr, const unsigned short* __restrict__ Ki,
    const unsigned short* __restrict__ Vtr, const unsigned short* __restrict__ Vti,
    const float* __restrict__ strength, const float* __restrict__ temp,
    unsigned short* __restrict__ Or, unsigned short* __restrict__ Oi) {
  __shared__ __align__(16) unsigned short pTile[8][16][32];
  int bh = blockIdx.y;
  int b = bh >> 4, h = bh & 15;
  int wave = threadIdx.x >> 5, lane = threadIdx.x & 31;
  int q0 = blockIdx.x * 128 + wave * 16;
  const size_t sd = (size_t)kS * kHD;
  const unsigned short* Qrb = Qr + (size_t)bh * sd;
  const unsigned short* Qib = Qi + (size_t)bh * sd;
  const unsigned short* Krb = Kr + (size_t)bh * sd;
  const unsigned short* Kib = Ki + (size_t)bh * sd;
  const unsigned short* Vrb = Vtr + (size_t)bh * sd;   // [D][S]
  const unsigned short* Vib = Vti + (size_t)bh * sd;

  float g   = 1.f / (1.f + __expf(-strength[0]));
  float tt  = fmaxf(temp[0], 0.01f);
  float lsc = g / tt;
  const float qksc = 0.125f;                           // 1/sqrt(64)

  v16bf aqr[2], aqi[2], anqr[2];                       // query-stationary frags
#pragma unroll
  for (int kh = 0; kh < 2; ++kh) {
    aqr[kh]  = load_a_frag(Qrb, q0, kh * 32, kHD);
    aqi[kh]  = load_a_frag(Qib, q0, kh * 32, kHD);
    anqr[kh] = neg_frag(aqr[kh]);
  }

  float m[8], l[8];
#pragma unroll
  for (int r = 0; r < 8; ++r) { m[r] = -__builtin_inff(); l[r] = 0.f; }
  v8f o_r[4] = {}, o_i[4] = {};
  int nCol = lane & 15, rowOff = (lane < 16) ? 0 : 8;

  for (int kk0 = 0; kk0 <= q0 + 15; kk0 += 32) {       // causal early exit
    v8f sr[2] = {}, si[2] = {};
#pragma unroll
    for (int sub = 0; sub < 2; ++sub)
#pragma unroll
      for (int kh = 0; kh < 2; ++kh) {
        v16bf bkr = load_b_frag(Krb, kk0 + sub * 16, kh * 32, kHD);
        v16bf bki = load_b_frag(Kib, kk0 + sub * 16, kh * 32, kHD);
        sr[sub] = wmma_bf16(aqr[kh],  bkr, sr[sub]);   // q_r.k_r + q_i.k_i
        sr[sub] = wmma_bf16(aqi[kh],  bki, sr[sub]);
        si[sub] = wmma_bf16(aqi[kh],  bkr, si[sub]);   // q_i.k_r - q_r.k_i
        si[sub] = wmma_bf16(anqr[kh], bki, si[sub]);
      }

    float lg[2][8], tmax[8];
#pragma unroll
    for (int r = 0; r < 8; ++r) {
      int q = q0 + r + rowOff;
      float rmax = -__builtin_inff();
#pragma unroll
      for (int sub = 0; sub < 2; ++sub) {
        float ar  = sr[sub][r] * qksc;
        float ai  = si[sub][r] * qksc;
        float mag = __fsqrt_rn(ar * ar + ai * ai + 1e-6f);
        int   kk  = kk0 + sub * 16 + nCol;
        float v   = (kk <= q) ? mag * lsc : -__builtin_inff();
        lg[sub][r] = v;
        rmax = fmaxf(rmax, v);
      }
#pragma unroll
      for (int off = 8; off >= 1; off >>= 1)           // 16-lane row reduce
        rmax = fmaxf(rmax, __shfl_xor(rmax, off, 32));
      tmax[r] = rmax;
    }

    float fs[8];
#pragma unroll
    for (int r = 0; r < 8; ++r) {
      float mn = fmaxf(m[r], tmax[r]);
      float f  = __expf(m[r] - mn);                    // exp(-inf)=0 first pass
      m[r] = mn; fs[r] = f;
      float p0 = __expf(lg[0][r] - mn);
      float p1 = __expf(lg[1][r] - mn);
      lg[0][r] = p0; lg[1][r] = p1;
      float rs = p0 + p1;
#pragma unroll
      for (int off = 8; off >= 1; off >>= 1)
        rs += __shfl_xor(rs, off, 32);
      l[r] = l[r] * f + rs;
    }
#pragma unroll
    for (int j = 0; j < 4; ++j)
#pragma unroll
      for (int r = 0; r < 8; ++r) { o_r[j][r] *= fs[r]; o_i[j][r] *= fs[r]; }

    // stage P (16x32) through LDS into A-fragment layout
#pragma unroll
    for (int sub = 0; sub < 2; ++sub)
#pragma unroll
      for (int r = 0; r < 8; ++r)
        pTile[wave][r + rowOff][sub * 16 + nCol] = f32_to_bf16(lg[sub][r]);
    asm volatile("s_wait_dscnt 0" ::: "memory");
    v16bf pa = load_a_frag(&pTile[wave][0][0], 0, 0, 32);
#pragma unroll
    for (int j = 0; j < 4; ++j) {
      v16bf bvr = load_b_frag(Vrb, j * 16, kk0, kS);
      v16bf bvi = load_b_frag(Vib, j * 16, kk0, kS);
      o_r[j] = wmma_bf16(pa, bvr, o_r[j]);
      o_i[j] = wmma_bf16(pa, bvi, o_i[j]);
    }
  }

  // normalize, store bf16 heads as [B,S,DIM] for the output projection
#pragma unroll
  for (int r = 0; r < 8; ++r) {
    int q = q0 + r + rowOff;
    float inv = 1.f / l[r];
    size_t ob = ((size_t)(b * kS + q)) * kDim + h * kHD;
#pragma unroll
    for (int j = 0; j < 4; ++j) {
      Or[ob + j * 16 + nCol] = f32_to_bf16(o_r[j][r] * inv);
      Oi[ob + j * 16 + nCol] = f32_to_bf16(o_i[j][r] * inv);
    }
  }
}

// ---------------------------------------------------------------------------
extern "C" void kernel_launch(void* const* d_in, const int* in_sizes, int n_in,
                              void* d_out, int out_size, void* d_ws, size_t ws_size,
                              hipStream_t stream) {
  (void)in_sizes; (void)n_in; (void)out_size; (void)ws_size;
  const float* real = (const float*)d_in[0];
  const float* imag = (const float*)d_in[1];
  const float* Wq_r = (const float*)d_in[2];  const float* bq_r = (const float*)d_in[3];
  const float* Wk_r = (const float*)d_in[4];  const float* bk_r = (const float*)d_in[5];
  const float* Wv_r = (const float*)d_in[6];  const float* bv_r = (const float*)d_in[7];
  const float* Wq_i = (const float*)d_in[8];  const float* bq_i = (const float*)d_in[9];
  const float* Wk_i = (const float*)d_in[10]; const float* bk_i = (const float*)d_in[11];
  const float* Wv_i = (const float*)d_in[12]; const float* bv_i = (const float*)d_in[13];
  const float* Wo_r = (const float*)d_in[14]; const float* bo_r = (const float*)d_in[15];
  const float* Wo_i = (const float*)d_in[16]; const float* bo_i = (const float*)d_in[17];
  const float* phase    = (const float*)d_in[18];
  const float* ent      = (const float*)d_in[19];
  const float* rotf     = (const float*)d_in[20];
  const float* strength = (const float*)d_in[21];
  const float* temp     = (const float*)d_in[22];

  char* ws = (char*)d_ws;
  size_t off = 0;
  auto carve = [&](size_t bytes) -> void* {
    void* p = ws + off;
    off += (bytes + 255) & ~(size_t)255;
    return p;
  };
  const size_t tokBF = (size_t)kTok * kDim * sizeof(unsigned short);
  const size_t tokF  = (size_t)kTok * kDim * sizeof(float);
  const size_t wBF   = (size_t)kDim * kDim * sizeof(unsigned short);

  unsigned short* Xr16 = (unsigned short*)carve(tokBF);
  unsigned short* Xi16 = (unsigned short*)carve(tokBF);
  unsigned short* WT[8];
  for (int i = 0; i < 8; ++i) WT[i] = (unsigned short*)carve(wBF);
  float* qr32 = (float*)carve(tokF);
  float* kr32 = (float*)carve(tokF);
  float* vr32 = (float*)carve(tokF);
  float* qi32 = (float*)carve(tokF);
  float* ki32 = (float*)carve(tokF);
  float* vi32 = (float*)carve(tokF);
  unsigned short* Qr16  = (unsigned short*)carve(tokBF);
  unsigned short* Qi16  = (unsigned short*)carve(tokBF);
  unsigned short* Kr16  = (unsigned short*)carve(tokBF);
  unsigned short* Ki16  = (unsigned short*)carve(tokBF);
  unsigned short* Vtr16 = (unsigned short*)carve(tokBF);
  unsigned short* Vti16 = (unsigned short*)carve(tokBF);
  unsigned short* Or16 = Xr16;   // reuse: X bf16 dead after the QKV GEMMs
  unsigned short* Oi16 = Xi16;

  // 1) activations -> bf16
  int nTok = kTok * kDim;
  cvt_kernel<<<(nTok + 255) / 256, 256, 0, stream>>>(real, Xr16, nTok);
  cvt_kernel<<<(nTok + 255) / 256, 256, 0, stream>>>(imag, Xi16, nTok);

  // 2) weights -> transposed bf16
  dim3 tg(kDim / 32, kDim / 32);
  wtrans_kernel<<<tg, 256, 0, stream>>>(Wq_r, WT[0]);
  wtrans_kernel<<<tg, 256, 0, stream>>>(Wk_r, WT[1]);
  wtrans_kernel<<<tg, 256, 0, stream>>>(Wv_r, WT[2]);
  wtrans_kernel<<<tg, 256, 0, stream>>>(Wq_i, WT[3]);
  wtrans_kernel<<<tg, 256, 0, stream>>>(Wk_i, WT[4]);
  wtrans_kernel<<<tg, 256, 0, stream>>>(Wv_i, WT[5]);
  wtrans_kernel<<<tg, 256, 0, stream>>>(Wo_r, WT[6]);
  wtrans_kernel<<<tg, 256, 0, stream>>>(Wo_i, WT[7]);

  // 3) QKV projections (WMMA)
  dim3 gg(kTok / 128, kDim / 128);
  gemm_bf16_kernel<<<gg, 256, 0, stream>>>(Xr16, WT[0], bq_r, qr32, kTok, kDim, kDim);
  gemm_bf16_kernel<<<gg, 256, 0, stream>>>(Xr16, WT[1], bk_r, kr32, kTok, kDim, kDim);
  gemm_bf16_kernel<<<gg, 256, 0, stream>>>(Xr16, WT[2], bv_r, vr32, kTok, kDim, kDim);
  gemm_bf16_kernel<<<gg, 256, 0, stream>>>(Xi16, WT[3], bq_i, qi32, kTok, kDim, kDim);
  gemm_bf16_kernel<<<gg, 256, 0, stream>>>(Xi16, WT[4], bk_i, ki32, kTok, kDim, kDim);
  gemm_bf16_kernel<<<gg, 256, 0, stream>>>(Xi16, WT[5], bv_i, vi32, kTok, kDim, kDim);

  // 4) V -> bf16 [B,H,D,S]
  v_transpose_kernel<<<(nTok + 255) / 256, 256, 0, stream>>>(vr32, Vtr16);
  v_transpose_kernel<<<(nTok + 255) / 256, 256, 0, stream>>>(vi32, Vti16);

  // 5) RoPE + ent mix + phase -> bf16 Q/K [B,H,S,D]
  int nR = kB * kS * (kHD / 2);
  rope_mix_phase_kernel<<<(nR + 255) / 256, 256, 0, stream>>>(
      qr32, qi32, kr32, ki32, phase, ent, rotf, Qr16, Qi16, Kr16, Ki16);

  // 6) complex-magnitude causal flash attention (WMMA)
  dim3 ag(kS / 128, kB * kH);
  attention_kernel<<<ag, 256, 0, stream>>>(Qr16, Qi16, Kr16, Ki16, Vtr16, Vti16,
                                           strength, temp, Or16, Oi16);

  // 7) output projections -> d_out (fp32), real then imag
  float* outR = (float*)d_out;
  float* outI = outR + (size_t)kB * kS * kDim;
  gemm_bf16_kernel<<<gg, 256, 0, stream>>>(Or16, WT[6], bo_r, outR, kTok, kDim, kDim);
  gemm_bf16_kernel<<<gg, 256, 0, stream>>>(Oi16, WT[7], bo_i, outI, kTok, kDim, kDim);
}
